// Block_softmoe_49572512531127
// MI455X (gfx1250) — matrix operand order repaired and checked by reading, not compiled, not run on hardware
//
#include <hip/hip_runtime.h>
#include <hip/hip_bf16.h>
#include <stdint.h>

// ---------------------------------------------------------------------------
// CDNA5 / gfx1250 transformer block with LoRA, built on v_wmma_f32_16x16x32_f16
// wave32 fragment layouts per CDNA5 ISA:
//   A (16x32 f16):  lane l holds row M = l&15; halves 0..7 -> K = kb+0..7,
//                   halves 8..15 -> K = kb+16..23, kb = (l>=16)?8:0
//   B (32x16 f16):  lane l holds col N = l&15; halves 0..7 -> K = kb+0..7,
//                   halves 8..15 -> K = kb+16..23, kb as for A   (column-major fragment; weights are
//                   pre-transposed to [N][K] so loads are contiguous b128s)
//   C/D (16x16 f32): element r of lane l -> row r + 8*(l>>4), col l&15
//
// GEMM: 32(M) x 64(N) register tile per wave (8 WMMAs / K=32 step); the B
// tile shared by all 8 waves of a block is staged into LDS with double-
// buffered GLOBAL_LOAD_ASYNC_TO_LDS_B128 (ASYNCcnt / s_wait_asynccnt), per
// CDNA5 ISA.
// ---------------------------------------------------------------------------

typedef _Float16 v16h __attribute__((ext_vector_type(16)));
typedef float    v8f  __attribute__((ext_vector_type(8)));
typedef _Float16 h8v  __attribute__((ext_vector_type(8)));

__device__ __forceinline__ v8f zero8() {
  v8f z;
#pragma unroll
  for (int i = 0; i < 8; ++i) z[i] = 0.0f;
  return z;
}

__device__ __forceinline__ v16h load_frag_a(const _Float16* __restrict__ p,
                                            int stride, int row0, int k0, int lane) {
  const int row = row0 + (lane & 15);
  const int kb  = (lane >> 4) << 3;   // 0 or 8
  const _Float16* r = p + (size_t)row * stride + k0 + kb;
  union { v16h v; h8v h[2]; } f;
  f.h[0] = *(const h8v*)(r);          // K = kb .. kb+7
  f.h[1] = *(const h8v*)(r + 16);     // K = kb+16 .. kb+23
  return f.v;
}

__device__ __forceinline__ v16h load_frag_b(const _Float16* p,
                                            int stride, int col0, int k0, int lane) {
  const int row = col0 + (lane & 15); // column of B == row of pre-transposed BT
  const int kb  = (lane >> 4) << 3;   // 0 or 8
  const _Float16* r = p + (size_t)row * stride + k0 + kb;
  union { v16h v; h8v h[2]; } f;
  f.h[0] = *(const h8v*)(r);          // K = kb .. kb+7
  f.h[1] = *(const h8v*)(r + 16);     // K = kb+16 .. kb+23
  return f.v;
}

__device__ __forceinline__ v8f wmma32(v16h a, v16h b, v8f c) {
  return __builtin_amdgcn_wmma_f32_16x16x32_f16(false, a, false, b,
                                                (short)0, c, false, false);
}

// Async global -> LDS copy of 16 bytes (per lane); tracked by ASYNCcnt.
__device__ __forceinline__ void async_b128_to_lds(uint32_t lds_off, const void* g) {
  asm volatile("global_load_async_to_lds_b128 %0, %1, off"
               :: "v"(lds_off), "v"((uint64_t)(uintptr_t)g)
               : "memory");
}

__device__ __forceinline__ float redmax16(float v) {
#pragma unroll
  for (int m = 8; m; m >>= 1) v = fmaxf(v, __shfl_xor(v, m, 16));
  return v;
}
__device__ __forceinline__ float redsum16(float v) {
#pragma unroll
  for (int m = 8; m; m >>= 1) v += __shfl_xor(v, m, 16);
  return v;
}

// --------------------------- elementwise prep ------------------------------

__global__ void cvt_f2h(const float* __restrict__ src, _Float16* __restrict__ dst, int n) {
  int i = blockIdx.x * 256 + threadIdx.x;
  if (i < n) dst[i] = (_Float16)src[i];
}

// src [K][N] f32 (row major)  ->  dst [N][K] f16 (row major)
__global__ void transpose_cvt(const float* __restrict__ src, _Float16* __restrict__ dst,
                              int K, int N) {
  int idx = blockIdx.x * 256 + threadIdx.x;
  if (idx >= K * N) return;
  int k = idx % K;
  int n = idx / K;
  dst[idx] = (_Float16)src[(size_t)k * N + n];
}

// Build V^T: vT[(bh*64 + d)*512 + s] = qkv[(b*512+s)*3072 + 2048 + h*64 + d]
__global__ void vt_prep(const _Float16* __restrict__ qkv, _Float16* __restrict__ vT) {
  int idx = blockIdx.x * 256 + threadIdx.x;       // total 8*16*64*512
  int s  = idx & 511;
  int d  = (idx >> 9) & 63;
  int bh = idx >> 15;
  int b  = bh >> 4;
  int h  = bh & 15;
  vT[idx] = qkv[((size_t)(b * 512 + s)) * 3072 + 2048 + h * 64 + d];
}

// ------------------------------- GEMM --------------------------------------
// OUT[M,N] = A[M,K] @ BT[N,K]^T  (+ LA[M,32] @ LBT[N,32]^T)  (+bias) (+resid)
// block = 256 threads = 8 waves; wave w -> 32-row strip, NT 16-col tiles.
// B tile (16*NT rows x 32 halves) staged in LDS via async loads, 2 stages.
template <int NT, bool LORA, bool BIAS, bool DO_GELU, bool RES, bool OUTF, bool OUTH>
__global__ __launch_bounds__(256) void gemm_wmma_kern(
    const _Float16* __restrict__ A, const _Float16* __restrict__ BT,
    const _Float16* __restrict__ LA, const _Float16* __restrict__ LBT,
    const float* __restrict__ bias, const _Float16* __restrict__ resid,
    float* __restrict__ outF, _Float16* __restrict__ outH,
    int M, int N, int K) {
  const int tid  = threadIdx.x;
  const int lane = tid & 31;
  const int wave = tid >> 5;
  const int m0 = (blockIdx.y * 8 + wave) * 32;   // 32 rows per wave
  const int n0 = blockIdx.x * (16 * NT);

  __shared__ __align__(16) _Float16 Bs[2][16 * NT][32];

  // Stage the (16*NT x 32) B tile for k0 into LDS buffer `stage`.
  const int chunks = (16 * NT * 32) / 8;          // one b128 per thread
  auto stage_issue = [&](int stage, int k0) {
    if (tid < chunks) {
      const int row = tid >> 2;                   // 4 x b128 per 32-half row
      const int col = (tid & 3) * 8;
      async_b128_to_lds((uint32_t)(uintptr_t)&Bs[stage][row][col],
                        BT + (size_t)(n0 + row) * K + k0 + col);
    }
  };

  v8f acc[2][NT];
#pragma unroll
  for (int i = 0; i < 2; ++i)
#pragma unroll
    for (int j = 0; j < NT; ++j) acc[i][j] = zero8();

  stage_issue(0, 0);
  for (int k0 = 0; k0 < K; k0 += 32) {
    const int stage = (k0 >> 5) & 1;
    const bool has_next = (k0 + 32 < K);
    if (has_next) {
      stage_issue(stage ^ 1, k0 + 32);
      __builtin_prefetch(A + (size_t)(m0 + (lane & 15)) * K + k0 + 32, 0, 1);
      asm volatile("s_wait_asynccnt 0x1" ::: "memory");
    } else {
      asm volatile("s_wait_asynccnt 0x0" ::: "memory");
    }
    __syncthreads();                              // tile `stage` visible to all

    v16h a0 = load_frag_a(A, K, m0,      k0, lane);
    v16h a1 = load_frag_a(A, K, m0 + 16, k0, lane);
    const _Float16* bs = &Bs[stage][0][0];
#pragma unroll
    for (int j = 0; j < NT; ++j) {
      v16h b = load_frag_b(bs, 32, 16 * j, 0, lane);
      acc[0][j] = wmma32(a0, b, acc[0][j]);
      acc[1][j] = wmma32(a1, b, acc[1][j]);
    }
    __syncthreads();                              // done reading before reuse
  }

  if (LORA) {  // extra K=32 step: rank-32 adapter (tiny; direct from global)
    v16h a0 = load_frag_a(LA, 32, m0,      0, lane);
    v16h a1 = load_frag_a(LA, 32, m0 + 16, 0, lane);
#pragma unroll
    for (int j = 0; j < NT; ++j) {
      v16h b = load_frag_b(LBT, 32, n0 + 16 * j, 0, lane);
      acc[0][j] = wmma32(a0, b, acc[0][j]);
      acc[1][j] = wmma32(a1, b, acc[1][j]);
    }
  }

  const int cl = lane & 15;
#pragma unroll
  for (int i = 0; i < 2; ++i) {
    const int r0 = m0 + 16 * i + ((lane >> 4) << 3);
#pragma unroll
    for (int j = 0; j < NT; ++j) {
      const int n = n0 + 16 * j + cl;
      const float bv = BIAS ? bias[n] : 0.0f;
#pragma unroll
      for (int r = 0; r < 8; ++r) {
        const size_t off = (size_t)(r0 + r) * N + n;
        float v = acc[i][j][r] + bv;
        if (RES)     v += (float)resid[off];
        if (DO_GELU) v = 0.5f * v * (1.0f + erff(v * 0.70710678118654752f));
        if (OUTF) outF[off] = v;
        if (OUTH) outH[off] = (_Float16)v;
      }
    }
  }
}

// ---------------------------- attention ------------------------------------
// One wave = 16-query strip for one (b,h). Online (flash) softmax over key
// chunks of 32: 4 WMMAs QK^T + 4 WMMAs P*V per chunk.
__global__ __launch_bounds__(128) void attn_kern(
    const _Float16* __restrict__ qkv, const _Float16* __restrict__ vT,
    const uint8_t* __restrict__ mask, _Float16* __restrict__ xo) {
  const int S = 512, C = 1024, N3 = 3072, HD = 64;
  const int lane = threadIdx.x & 31;
  const int wave = threadIdx.x >> 5;
  const int bh = blockIdx.y;
  const int b = bh >> 4, h = bh & 15;
  const int q0 = blockIdx.x * 64 + wave * 16;
  const int hi = lane >> 4;
  const int cl = lane & 15;

  const _Float16* qb  = qkv + (size_t)b * S * N3 + h * HD;  // q cols
  const _Float16* kb  = qb + C;                             // k cols
  const _Float16* vtb = vT + (size_t)bh * HD * S;           // [d][s]
  const uint8_t*  mrow = mask + b * 3 * S;

  v16h qa0 = load_frag_a(qb, N3, q0, 0, lane);
  v16h qa1 = load_frag_a(qb, N3, q0, 32, lane);

  float rm[8], rs[8];
#pragma unroll
  for (int r = 0; r < 8; ++r) { rm[r] = -__builtin_inff(); rs[r] = 0.0f; }
  v8f o0 = zero8(), o1 = zero8(), o2 = zero8(), o3 = zero8();

  __shared__ __align__(16) _Float16 Pt[4][16][32];

  for (int kt = 0; kt < S; kt += 32) {
    v8f s0 = zero8(), s1 = zero8();
    s0 = wmma32(qa0, load_frag_b(kb, N3, kt,       0, lane), s0);
    s0 = wmma32(qa1, load_frag_b(kb, N3, kt,      32, lane), s0);
    s1 = wmma32(qa0, load_frag_b(kb, N3, kt + 16,  0, lane), s1);
    s1 = wmma32(qa1, load_frag_b(kb, N3, kt + 16, 32, lane), s1);

    const bool mk0 = mrow[kt + cl] != 0;
    const bool mk1 = mrow[kt + 16 + cl] != 0;

    float alpha[8];
#pragma unroll
    for (int r = 0; r < 8; ++r) {
      float a0 = mk0 ? s0[r] * 0.125f : -__builtin_inff();
      float a1 = mk1 ? s1[r] * 0.125f : -__builtin_inff();
      float mt = redmax16(fmaxf(a0, a1));
      float nm = fmaxf(rm[r], mt);
      float al, e0, e1;
      if (nm == -__builtin_inff()) {      // all keys masked so far
        al = 1.0f; e0 = 0.0f; e1 = 0.0f;
      } else {
        al = __expf(rm[r] - nm);          // rm==-inf -> 0
        e0 = (a0 == -__builtin_inff()) ? 0.0f : __expf(a0 - nm);
        e1 = (a1 == -__builtin_inff()) ? 0.0f : __expf(a1 - nm);
      }
      rs[r] = rs[r] * al + redsum16(e0 + e1);
      rm[r] = nm;
      alpha[r] = al;
      Pt[wave][r + 8 * hi][cl]      = (_Float16)e0;
      Pt[wave][r + 8 * hi][16 + cl] = (_Float16)e1;
    }
#pragma unroll
    for (int r = 0; r < 8; ++r) {
      o0[r] *= alpha[r]; o1[r] *= alpha[r];
      o2[r] *= alpha[r]; o3[r] *= alpha[r];
    }
    __syncthreads();   // LDS stores -> A-fragment reads (cross-lane)
    v16h pa = load_frag_a(&Pt[wave][0][0], 32, 0, 0, lane);
    o0 = wmma32(pa, load_frag_b(vtb, S,  0, kt, lane), o0);
    o1 = wmma32(pa, load_frag_b(vtb, S, 16, kt, lane), o1);
    o2 = wmma32(pa, load_frag_b(vtb, S, 32, kt, lane), o2);
    o3 = wmma32(pa, load_frag_b(vtb, S, 48, kt, lane), o3);
    __syncthreads();
  }

  _Float16* xob = xo + (size_t)(b * S + q0) * C + h * HD;
#pragma unroll
  for (int r = 0; r < 8; ++r) {
    const int qr = r + 8 * hi;
    const float inv = (rs[r] > 0.0f) ? 1.0f / rs[r] : 0.0f;  // NaN guard
    xob[(size_t)qr * C +  0 + cl] = (_Float16)(o0[r] * inv);
    xob[(size_t)qr * C + 16 + cl] = (_Float16)(o1[r] * inv);
    xob[(size_t)qr * C + 32 + cl] = (_Float16)(o2[r] * inv);
    xob[(size_t)qr * C + 48 + cl] = (_Float16)(o3[r] * inv);
  }
}

// ------------------------------- driver ------------------------------------

extern "C" void kernel_launch(void* const* d_in, const int* in_sizes, int n_in,
                              void* d_out, int out_size, void* d_ws, size_t ws_size,
                              hipStream_t stream) {
  const int B = 8, S = 512, C = 1024, R = 32, N3 = 3072, M = B * S;

  const float*   x       = (const float*)d_in[0];
  const uint8_t* mask    = (const uint8_t*)d_in[1];  // jax bool -> 1 byte/elem
  const float*   qkv_w   = (const float*)d_in[2];
  const float*   qkv_la  = (const float*)d_in[3];
  const float*   qkv_lb  = (const float*)d_in[4];
  const float*   proj_w  = (const float*)d_in[5];
  const float*   proj_b  = (const float*)d_in[6];
  const float*   proj_la = (const float*)d_in[7];
  const float*   proj_lb = (const float*)d_in[8];
  const float*   fc1_w   = (const float*)d_in[9];
  const float*   fc1_b   = (const float*)d_in[10];
  const float*   fc1_la  = (const float*)d_in[11];
  const float*   fc1_lb  = (const float*)d_in[12];
  const float*   fc2_w   = (const float*)d_in[13];
  const float*   fc2_b   = (const float*)d_in[14];
  const float*   fc2_la  = (const float*)d_in[15];
  const float*   fc2_lb  = (const float*)d_in[16];

  // carve workspace (f16 regions, 256B aligned)
  char* p = (char*)d_ws;
  auto alloc = [&](size_t elems) -> _Float16* {
    _Float16* r = (_Float16*)p;
    p += (elems * sizeof(_Float16) + 255) & ~(size_t)255;
    return r;
  };
  _Float16* xh     = alloc((size_t)M * C);
  _Float16* wqkvT  = alloc((size_t)N3 * C);
  _Float16* laqT   = alloc((size_t)R * C);
  _Float16* lbqT   = alloc((size_t)N3 * R);
  _Float16* wprojT = alloc((size_t)C * C);
  _Float16* lapT   = alloc((size_t)R * C);
  _Float16* lbpT   = alloc((size_t)C * R);
  _Float16* wfc1T  = alloc((size_t)C * C);
  _Float16* la1T   = alloc((size_t)R * C);
  _Float16* lb1T   = alloc((size_t)C * R);
  _Float16* wfc2T  = alloc((size_t)C * C);
  _Float16* la2T   = alloc((size_t)R * C);
  _Float16* lb2T   = alloc((size_t)C * R);
  _Float16* tbuf   = alloc((size_t)M * R);
  _Float16* qkvh   = alloc((size_t)M * N3);
  _Float16* vT     = alloc((size_t)B * 16 * 64 * S);
  _Float16* xoh    = alloc((size_t)M * C);
  _Float16* h1     = alloc((size_t)M * C);
  _Float16* xo2    = alloc((size_t)M * C);

  { int n = M * C; cvt_f2h<<<(n + 255) / 256, 256, 0, stream>>>(x, xh, n); }
  auto tr = [&](const float* src, _Float16* dst, int K, int N) {
    int n = K * N;
    transpose_cvt<<<(n + 255) / 256, 256, 0, stream>>>(src, dst, K, N);
  };
  tr(qkv_w, wqkvT, C, N3);  tr(qkv_la, laqT, C, R);   tr(qkv_lb, lbqT, R, N3);
  tr(proj_w, wprojT, C, C); tr(proj_la, lapT, C, R);  tr(proj_lb, lbpT, R, C);
  tr(fc1_w, wfc1T, C, C);   tr(fc1_la, la1T, C, R);   tr(fc1_lb, lb1T, R, C);
  tr(fc2_w, wfc2T, C, C);   tr(fc2_la, la2T, C, R);   tr(fc2_lb, lb2T, R, C);

  dim3 blk(256);
  // t = x16 @ qkv_la ; qkv = x16 @ qkv_w + t @ qkv_lb
  gemm_wmma_kern<2, false, false, false, false, false, true>
      <<<dim3(1, M / 256), blk, 0, stream>>>(xh, laqT, nullptr, nullptr,
          nullptr, nullptr, nullptr, tbuf, M, R, C);
  gemm_wmma_kern<4, true, false, false, false, false, true>
      <<<dim3(N3 / 64, M / 256), blk, 0, stream>>>(xh, wqkvT, tbuf, lbqT,
          nullptr, nullptr, nullptr, qkvh, M, N3, C);

  { int n = B * 16 * 64 * S; vt_prep<<<(n + 255) / 256, 256, 0, stream>>>(qkvh, vT); }
  attn_kern<<<dim3(S / 64, B * 16), dim3(128), 0, stream>>>(qkvh, vT, mask, xoh);

  // h = gelu(xo @ fc1_w + b + lora)
  gemm_wmma_kern<2, false, false, false, false, false, true>
      <<<dim3(1, M / 256), blk, 0, stream>>>(xoh, la1T, nullptr, nullptr,
          nullptr, nullptr, nullptr, tbuf, M, R, C);
  gemm_wmma_kern<4, true, true, true, false, false, true>
      <<<dim3(C / 64, M / 256), blk, 0, stream>>>(xoh, wfc1T, tbuf, lb1T,
          fc1_b, nullptr, nullptr, h1, M, C, C);

  // xo2 = xo + (h @ fc2_w + b + lora)
  gemm_wmma_kern<2, false, false, false, false, false, true>
      <<<dim3(1, M / 256), blk, 0, stream>>>(h1, la2T, nullptr, nullptr,
          nullptr, nullptr, nullptr, tbuf, M, R, C);
  gemm_wmma_kern<4, true, true, false, true, false, true>
      <<<dim3(C / 64, M / 256), blk, 0, stream>>>(h1, wfc2T, tbuf, lb2T,
          fc2_b, xoh, nullptr, xo2, M, C, C);

  // out = xo2 @ proj_w + b + lora   (f32 straight to d_out)
  gemm_wmma_kern<2, false, false, false, false, false, true>
      <<<dim3(1, M / 256), blk, 0, stream>>>(xo2, lapT, nullptr, nullptr,
          nullptr, nullptr, nullptr, tbuf, M, R, C);
  gemm_wmma_kern<4, true, true, false, false, true, false>
      <<<dim3(C / 64, M / 256), blk, 0, stream>>>(xo2, wprojT, tbuf, lbpT,
          proj_b, nullptr, (float*)d_out, nullptr, M, C, C);
}
